// AttnBlock_16381005267501
// MI455X (gfx1250) — matrix operand order repaired and checked by reading, not compiled, not run on hardware
//
#include <hip/hip_runtime.h>

// ---------------------------------------------------------------------------
// Types / helpers
// ---------------------------------------------------------------------------
typedef __attribute__((ext_vector_type(16))) __bf16 bf16x16;
typedef __attribute__((ext_vector_type(8)))  float   f32x8;
typedef __attribute__((ext_vector_type(4)))  float   f32x4;
typedef __attribute__((ext_vector_type(4)))  unsigned int u32x4;
typedef __attribute__((ext_vector_type(2)))  unsigned int u32x2;

static __device__ __forceinline__ __bf16 f2bf(float f) {
    unsigned u = __builtin_bit_cast(unsigned, f);
    unsigned r = (u + 0x7FFFu + ((u >> 16) & 1u)) >> 16;   // RNE
    unsigned short s = (unsigned short)r;
    return __builtin_bit_cast(__bf16, s);
}
static __device__ __forceinline__ float bf2f(__bf16 h) {
    unsigned short s = __builtin_bit_cast(unsigned short, h);
    unsigned u = ((unsigned)s) << 16;
    return __builtin_bit_cast(float, u);
}

union AFrag { bf16x16 v; u32x4 q[2]; };
union Pack8 { __bf16 h[8]; u32x4 u; };
union Pack4 { __bf16 h[4]; u32x2 u; };

#define WMMA_BF16(a, b, c) \
    __builtin_amdgcn_wmma_f32_16x16x32_bf16(false, (a), false, (b), (short)0, (c), false, false)

// Problem constants
#define BB   4
#define SS   2048
#define HH   1024

// ---------------------------------------------------------------------------
// Epilogue helpers: per-wave LDS transpose for coalesced row-major stores
// ---------------------------------------------------------------------------
static __device__ __forceinline__ void epi_row_bf16(
    const f32x8 (&c)[2][4], char* smembase, int wid, int l,
    __bf16* dst, int grow0, int gcol0, int ldc, const float* biasp)
{
    __bf16* scr = (__bf16*)(smembase + wid * 4608);  // 32 rows x 72 (pad) bf16
    int l15 = l & 15, hi = l >> 4;
#pragma unroll
    for (int j = 0; j < 4; ++j) {
        float bj = biasp ? biasp[j * 16 + l15] : 0.0f;
#pragma unroll
        for (int i2 = 0; i2 < 2; ++i2)
#pragma unroll
            for (int e = 0; e < 8; ++e)
                scr[(i2 * 16 + hi * 8 + e) * 72 + j * 16 + l15] = f2bf(c[i2][j][e] + bj);
    }
    __bf16* drow = dst + (size_t)(grow0 + l) * ldc + gcol0;
#pragma unroll
    for (int i = 0; i < 8; ++i)
        *(u32x4*)(drow + 8 * i) = *(const u32x4*)(scr + l * 72 + 8 * i);
}

static __device__ __forceinline__ void epi_row_f32(
    const f32x8 (&c)[2][4], char* smembase, int wid, int l,
    float* dst, int grow0, int gcol0, int ldc, const float* biasp)
{
    float* scr = (float*)(smembase + wid * 8704);    // 32 rows x 68 (pad) f32
    int l15 = l & 15, hi = l >> 4;
#pragma unroll
    for (int j = 0; j < 4; ++j) {
        float bj = biasp ? biasp[j * 16 + l15] : 0.0f;
#pragma unroll
        for (int i2 = 0; i2 < 2; ++i2)
#pragma unroll
            for (int e = 0; e < 8; ++e)
                scr[(i2 * 16 + hi * 8 + e) * 68 + j * 16 + l15] = c[i2][j][e] + bj;
    }
    float* drow = dst + (size_t)(grow0 + l) * ldc + gcol0;
#pragma unroll
    for (int i = 0; i < 16; ++i)
        *(f32x4*)(drow + 4 * i) = *(const f32x4*)(scr + l * 68 + 4 * i);
}

// ---------------------------------------------------------------------------
// Generic tiled bf16 WMMA GEMM: C[128x128] per block, 8 waves of 32x64.
//   AL: A-tile source  (fp32 row-major / bf16 row-major / softmax(S^T) fused)
//   EP: epilogue       (QKV split / bf16 row-major / f32 row-major)
// Register-staged double-buffered LDS pipeline: global loads for tile i+1 are
// issued into VGPRs while WMMAs consume tile i; one s_barrier per K-step.
// LDS tile row stride 56 elems (112B): 16B-aligned b128 frag loads, 16 distinct
// banks across the 16-lane groups.
// ---------------------------------------------------------------------------
enum { AL_F32 = 0, AL_BF16 = 1, AL_SOFTMAX = 2 };
enum { EP_QKV = 0, EP_BF16R = 1, EP_F32R = 2 };

template <int AL, int EP>
__global__ __launch_bounds__(256, 1) void gemm_kernel(
    const void* __restrict__ Abase, const __bf16* __restrict__ Wt,
    const float* __restrict__ bias,
    void* __restrict__ out0, __bf16* __restrict__ outK, __bf16* __restrict__ outV,
    const float* __restrict__ smaxp, const float* __restrict__ srcpp,
    int KK, int aRowStride, int ldc,
    unsigned long long aBatch, unsigned long long wBatch,
    unsigned long long oBatch, unsigned long long statBatch)
{
    constexpr int LDT   = 56;
    constexpr int TILEB = 14336;                     // 128 x 56 bf16
    constexpr int SMEM  = (EP == EP_F32R) ? 69632 : 57344;
    __shared__ __attribute__((aligned(16))) char smem[SMEM];
    __bf16* As0 = (__bf16*)(smem);
    __bf16* Bs0 = (__bf16*)(smem + TILEB);
    __bf16* As1 = (__bf16*)(smem + 2 * TILEB);
    __bf16* Bs1 = (__bf16*)(smem + 3 * TILEB);

    const int tid = threadIdx.x;
    const int wid = tid >> 5;
    const int l   = tid & 31;
    const int l15 = l & 15;
    const int hi  = l >> 4;
    const int gm0 = blockIdx.y * 128;
    const int gn0 = blockIdx.x * 128;
    const int bz  = blockIdx.z;
    const int wm  = (wid >> 1) * 32;
    const int wn  = (wid & 1) * 64;

    const __bf16* Wtb = Wt + (size_t)bz * wBatch;

    // loader thread mappings
    const int ldr = tid >> 1;           // 0..127 (row, A/B bf16+f32 loaders)
    const int ldo = (tid & 1) * 16;     // element offset 0/16
    const int skr = tid >> 3;           // 0..31  (k-row, softmax loader)
    const int sqo = (tid & 7) * 16;     // q offset 0..112

    // hoisted softmax stats (per 16 q handled by this thread's loader slice)
    float qmax[16], qrcp[16];
    if constexpr (AL == AL_SOFTMAX) {
        const float* mx = smaxp + (size_t)bz * statBatch + gm0 + sqo;
        const float* rc = srcpp + (size_t)bz * statBatch + gm0 + sqo;
#pragma unroll
        for (int i = 0; i < 4; ++i) {
            f32x4 a = *(const f32x4*)(mx + 4 * i);
            f32x4 b = *(const f32x4*)(rc + 4 * i);
#pragma unroll
            for (int j = 0; j < 4; ++j) { qmax[4 * i + j] = a[j]; qrcp[4 * i + j] = b[j]; }
        }
    }

    f32x8 c[2][4] = {};

    // staged registers for the in-flight tile
    f32x4 raf[4];        // AL_F32
    u32x4 ra0, ra1;      // AL_BF16 (ra0) / AL_SOFTMAX (ra0, ra1)
    u32x4 rb;            // B tile slice

    auto gload = [&](int kt) {
        if constexpr (AL == AL_F32) {
            const float* src = (const float*)Abase + (size_t)(gm0 + ldr) * aRowStride + kt + ldo;
#pragma unroll
            for (int i = 0; i < 4; ++i) raf[i] = *(const f32x4*)(src + 4 * i);
            if (kt + 32 < KK) __builtin_prefetch(src + 32, 0, 1);
        } else if constexpr (AL == AL_BF16) {
            const __bf16* src = (const __bf16*)Abase + (size_t)bz * aBatch +
                                (size_t)(gm0 + ldr) * aRowStride + kt + ldo;
            ra0 = *(const u32x4*)src;
            if (kt + 32 < KK) __builtin_prefetch(src + 32, 0, 1);
        } else {  // AL_SOFTMAX: raw bf16 logits from S^T[k][q]
            const __bf16* src = (const __bf16*)Abase + (size_t)bz * aBatch +
                                (size_t)(kt + skr) * aRowStride + gm0 + sqo;
            ra0 = *(const u32x4*)src;
            ra1 = *(const u32x4*)(src + 8);
            if (kt + 32 < KK) __builtin_prefetch(src + (size_t)32 * aRowStride, 0, 1);
        }
        const __bf16* bsrc = Wtb + (size_t)(gn0 + ldr) * KK + kt + ldo;
        rb = *(const u32x4*)bsrc;
        if (kt + 32 < KK) __builtin_prefetch(bsrc + 32, 0, 1);
    };

    auto dsstore = [&](__bf16* As, __bf16* Bs) {
        if constexpr (AL == AL_F32) {
#pragma unroll
            for (int i = 0; i < 4; ++i) {
                Pack4 p;
#pragma unroll
                for (int j = 0; j < 4; ++j) p.h[j] = f2bf(raf[i][j]);
                *(u32x2*)&As[ldr * LDT + ldo + 4 * i] = p.u;
            }
        } else if constexpr (AL == AL_BF16) {
            *(u32x4*)&As[ldr * LDT + ldo] = ra0;
        } else {  // softmax transform at store time: p = exp(v - m) * (1/sum)
            Pack8 w0, w1;
            w0.u = ra0;
            w1.u = ra1;
#pragma unroll
            for (int e = 0; e < 8; ++e)
                As[(sqo + e) * LDT + skr] =
                    f2bf(__expf(bf2f(w0.h[e]) - qmax[e]) * qrcp[e]);
#pragma unroll
            for (int e = 0; e < 8; ++e)
                As[(sqo + 8 + e) * LDT + skr] =
                    f2bf(__expf(bf2f(w1.h[e]) - qmax[8 + e]) * qrcp[8 + e]);
        }
        *(u32x4*)&Bs[ldr * LDT + ldo] = rb;
    };

    auto compute = [&](const __bf16* As, const __bf16* Bs) {
        AFrag a[2], bf[4];
#pragma unroll
        for (int i2 = 0; i2 < 2; ++i2) {
            const __bf16* p = &As[(wm + i2 * 16 + l15) * LDT + (hi ? 8 : 0)];
            a[i2].q[0] = *(const u32x4*)p;
            a[i2].q[1] = *(const u32x4*)(p + 16);
        }
#pragma unroll
        for (int j = 0; j < 4; ++j) {
            const __bf16* p = &Bs[(wn + j * 16 + l15) * LDT + (hi ? 8 : 0)];
            bf[j].q[0] = *(const u32x4*)p;
            bf[j].q[1] = *(const u32x4*)(p + 16);
        }
#pragma unroll
        for (int i2 = 0; i2 < 2; ++i2)
#pragma unroll
            for (int j = 0; j < 4; ++j)
                c[i2][j] = WMMA_BF16(a[i2].v, bf[j].v, c[i2][j]);
    };

    // ---- software-pipelined main loop: one barrier per K-step ----
    const int nst = KK >> 5;
    gload(0);
    for (int it = 0; it < nst; ++it) {
        __bf16* Asb = (it & 1) ? As1 : As0;
        __bf16* Bsb = (it & 1) ? Bs1 : Bs0;
        dsstore(Asb, Bsb);         // waits on the staged global loads
        __syncthreads();           // tile ready; also fences prior buffer reuse
        if (it + 1 < nst) gload((it + 1) << 5);   // overlap with WMMAs below
        compute(Asb, Bsb);
    }
    __syncthreads();               // before reusing smem as epilogue scratch

    // ---- epilogue ----
    if constexpr (EP == EP_QKV) {
        int sel = gn0 >> 10;  // 0=Q, 1=K, 2=V   (3072-wide N split)
        if (sel < 2) {
            __bf16* dst = sel ? outK : (__bf16*)out0;
            epi_row_bf16(c, smem, wid, l, dst, gm0 + wm, (gn0 - sel * 1024) + wn, ldc,
                         bias + gn0 + wn);
        } else {
            // V stored transposed: Vt[b][h][s]; 8 accumulator rows = contiguous s
#pragma unroll
            for (int i2 = 0; i2 < 2; ++i2)
#pragma unroll
                for (int j = 0; j < 4; ++j) {
                    int n = gn0 + wn + j * 16 + l15;
                    float bj = bias[n];
                    int mrow = gm0 + wm + i2 * 16 + hi * 8;
                    int bb = mrow >> 11;
                    int ss = mrow & (SS - 1);
                    Pack8 p;
#pragma unroll
                    for (int e = 0; e < 8; ++e) p.h[e] = f2bf(c[i2][j][e] + bj);
                    *(u32x4*)(outV + ((size_t)(bb * HH + (n - 2048))) * SS + ss) = p.u;
                }
        }
    } else if constexpr (EP == EP_BF16R) {
        __bf16* dst = (__bf16*)out0 + (size_t)bz * oBatch;
        epi_row_bf16(c, smem, wid, l, dst, gm0 + wm, gn0 + wn, ldc, nullptr);
    } else {
        float* dst = (float*)out0;
        epi_row_f32(c, smem, wid, l, dst, gm0 + wm, gn0 + wn, ldc, bias + gn0 + wn);
    }
}

// ---------------------------------------------------------------------------
// Attention stats: per wave = 16 query rows. Streams 32-key tiles, WMMA-based
// Q@K^T accumulation over H, online softmax stats (shfl-xor row reductions),
// stores raw bf16 logits column-major S^T[b][k][q] with packed b128 stores.
// ---------------------------------------------------------------------------
__global__ __launch_bounds__(256, 1) void attn_stats_kernel(
    const __bf16* __restrict__ Q, const __bf16* __restrict__ K,
    __bf16* __restrict__ St, float* __restrict__ smax, float* __restrict__ srcp)
{
    const int tid = threadIdx.x;
    const int wid = tid >> 5;
    const int l = tid & 31, l15 = l & 15, hi = l >> 4;
    const int gw = blockIdx.x * 8 + wid;
    const int b  = gw >> 7;            // 128 q-blocks of 16 per batch
    const int q0 = (gw & 127) << 4;
    const float scale = 0.03125f;      // 1/sqrt(1024)

    const __bf16* Qrow = Q + ((size_t)(b * SS + q0 + l15)) * HH + (hi ? 8 : 0);

    float mrun[8], srun[8];
#pragma unroll
    for (int e = 0; e < 8; ++e) { mrun[e] = -3.0e38f; srun[e] = 0.0f; }

    for (int nk = 0; nk < SS; nk += 32) {
        f32x8 c0 = {}, c1 = {};
        const __bf16* K0 = K + ((size_t)(b * SS + nk + l15)) * HH + (hi ? 8 : 0);
        const __bf16* K1 = K0 + (size_t)16 * HH;
        if (nk + 32 < SS) {            // pull next key tile into GL2
            __builtin_prefetch(K0 + (size_t)32 * HH, 0, 1);
            __builtin_prefetch(K1 + (size_t)32 * HH, 0, 1);
        }
        for (int kt = 0; kt < HH; kt += 32) {
            AFrag a, b0, b1;
            a.q[0]  = *(const u32x4*)(Qrow + kt);
            a.q[1]  = *(const u32x4*)(Qrow + kt + 16);
            b0.q[0] = *(const u32x4*)(K0 + kt);
            b0.q[1] = *(const u32x4*)(K0 + kt + 16);
            b1.q[0] = *(const u32x4*)(K1 + kt);
            b1.q[1] = *(const u32x4*)(K1 + kt + 16);
            c0 = WMMA_BF16(a.v, b0.v, c0);
            c1 = WMMA_BF16(a.v, b1.v, c1);
        }
        // bf16 round-trip so stats match the stored logits exactly
        float v0[8], v1[8];
#pragma unroll
        for (int e = 0; e < 8; ++e) {
            v0[e] = bf2f(f2bf(c0[e] * scale));
            v1[e] = bf2f(f2bf(c1[e] * scale));
        }
#pragma unroll
        for (int e = 0; e < 8; ++e) {
            float t = fmaxf(v0[e], v1[e]);
            t = fmaxf(t, __shfl_xor(t, 1, 32));
            t = fmaxf(t, __shfl_xor(t, 2, 32));
            t = fmaxf(t, __shfl_xor(t, 4, 32));
            t = fmaxf(t, __shfl_xor(t, 8, 32));
            float mn = fmaxf(mrun[e], t);
            float ps = __expf(v0[e] - mn) + __expf(v1[e] - mn);
            ps += __shfl_xor(ps, 1, 32);
            ps += __shfl_xor(ps, 2, 32);
            ps += __shfl_xor(ps, 4, 32);
            ps += __shfl_xor(ps, 8, 32);
            srun[e] = srun[e] * __expf(mrun[e] - mn) + ps;
            mrun[e] = mn;
        }
        Pack8 p0, p1;
#pragma unroll
        for (int e = 0; e < 8; ++e) { p0.h[e] = f2bf(v0[e]); p1.h[e] = f2bf(v1[e]); }
        __bf16* srow = St + ((size_t)(b * SS + nk + l15)) * SS + q0 + hi * 8;
        *(u32x4*)srow = p0.u;
        *(u32x4*)(srow + (size_t)16 * SS) = p1.u;
    }
    if (l15 == 0) {
        int base = b * SS + q0 + hi * 8;
#pragma unroll
        for (int e = 0; e < 8; ++e) {
            smax[base + e] = mrun[e];
            srcp[base + e] = 1.0f / srun[e];
        }
    }
}

// ---------------------------------------------------------------------------
// Weight transpose + fp32->bf16 convert:  in[K][N] f32  ->  out[N][K] bf16
// ---------------------------------------------------------------------------
__global__ __launch_bounds__(256, 1) void wt_transpose_kernel(
    const float* __restrict__ in, __bf16* __restrict__ out, int K, int N)
{
    __shared__ float t[32][33];
    int tx = threadIdx.x, ty = threadIdx.y;
    int n0 = blockIdx.x * 32, k0 = blockIdx.y * 32;
#pragma unroll
    for (int j = 0; j < 4; ++j)
        t[ty + 8 * j][tx] = in[(size_t)(k0 + ty + 8 * j) * N + n0 + tx];
    __syncthreads();
#pragma unroll
    for (int j = 0; j < 4; ++j)
        out[(size_t)(n0 + ty + 8 * j) * K + k0 + tx] = f2bf(t[tx][ty + 8 * j]);
}

// ---------------------------------------------------------------------------
// Launch
// ---------------------------------------------------------------------------
extern "C" void kernel_launch(void* const* d_in, const int* in_sizes, int n_in,
                              void* d_out, int out_size, void* d_ws, size_t ws_size,
                              hipStream_t stream)
{
    (void)in_sizes; (void)n_in; (void)out_size; (void)ws_size;

    const float* x      = (const float*)d_in[0];
    const float* w_attn = (const float*)d_in[1];
    const float* b_attn = (const float*)d_in[2];
    const float* w_proj = (const float*)d_in[3];
    const float* b_proj = (const float*)d_in[4];
    float* out = (float*)d_out;
    char* ws = (char*)d_ws;

    // workspace carve (bytes), all 16B aligned; total ~104 MB (fits L2)
    __bf16* Wa = (__bf16*)(ws + 0);          //  6291456  w_attn^T bf16 [3072][1024]
    __bf16* Wp = (__bf16*)(ws + 6291456);    //  2097152  w_proj^T bf16 [1024][1024]
    __bf16* Qb = (__bf16*)(ws + 8388608);    // 16777216  Q  [B*S][H]
    __bf16* Kb = (__bf16*)(ws + 25165824);   // 16777216  K  [B*S][H]
    __bf16* Vt = (__bf16*)(ws + 41943040);   // 16777216  V^T [B][H][S]
    __bf16* St = (__bf16*)(ws + 58720256);   // 33554432  S^T logits [B][k][q]
    __bf16* Ob = (__bf16*)(ws + 92274688);   // 16777216  O  [B*S][H]
    float*  Mx = (float*)(ws + 109051904);   //    32768  row max
    float*  Rs = (float*)(ws + 109084672);   //    32768  1/row sum

    dim3 blk(256);

    // 1) weight prep
    wt_transpose_kernel<<<dim3(96, 32), dim3(32, 8), 0, stream>>>(w_attn, Wa, 1024, 3072);
    wt_transpose_kernel<<<dim3(32, 32), dim3(32, 8), 0, stream>>>(w_proj, Wp, 1024, 1024);

    // 2) fused QKV projection (x fp32 converted in the loader) -> Q, K, V^T bf16
    gemm_kernel<AL_F32, EP_QKV><<<dim3(24, 64, 1), blk, 0, stream>>>(
        x, Wa, b_attn, Qb, Kb, Vt, nullptr, nullptr,
        1024, 1024, 1024, 0ull, 0ull, 0ull, 0ull);

    // 3) scores + online softmax stats, S^T logits
    attn_stats_kernel<<<dim3(64), blk, 0, stream>>>(Qb, Kb, St, Mx, Rs);

    // 4) O = softmax(S) @ V  (softmax fused into the A-tile loader)
    gemm_kernel<AL_SOFTMAX, EP_BF16R><<<dim3(8, 16, 4), blk, 0, stream>>>(
        St, Vt, nullptr, Ob, nullptr, nullptr, Mx, Rs,
        2048, 2048, 1024,
        (unsigned long long)SS * SS, (unsigned long long)HH * SS,
        (unsigned long long)SS * HH, (unsigned long long)SS);

    // 5) out = O @ w_proj + b_proj  (fp32 output)
    gemm_kernel<AL_BF16, EP_F32R><<<dim3(8, 64, 1), blk, 0, stream>>>(
        Ob, Wp, b_proj, out, nullptr, nullptr, nullptr, nullptr,
        1024, 1024, 1024, 0ull, 0ull, 0ull, 0ull);
}
